// AdaptiveReconstructionTransformer_16320875725480
// MI455X (gfx1250) — matrix operand order repaired and checked by reading, not compiled, not run on hardware
//
#include <hip/hip_runtime.h>
#include <cmath>

// MI455X / gfx1250, wave32. Matrix math via v_wmma_f32_16x16x32_f16 with
// LDS-tiled 64x64 macro-tiles and fragment-ready LDS layouts:
// each WMMA operand fragment is one contiguous 32B run -> 2x ds_load_b128.

typedef __attribute__((ext_vector_type(16))) _Float16 v16h;
typedef __attribute__((ext_vector_type(8)))  float    v8f;

// Permute K so that the A-fragment of each half-wave is contiguous:
// half 0 uses K {0..7,16..23} -> positions 0..15 ; half 1 uses {8..15,24..31}
// -> positions 16..31, in exact ISA 7.12.2 VGPR order.
__device__ __forceinline__ int aperm(int k) {
  return (((k >> 3) & 1) << 4) | (k & 7) | ((k >> 1) & 8);
}

// ---------------------------------------------------------------------------
// Generic strided batched GEMM, LDS-tiled, 64(M) x 64(N) per 128-thread block.
// Wave w computes rows [m0+16w, m0+16w+16) x 64 cols via 4 accumulators.
// Per-z (blockIdx.z) base offsets: off = (z/bdiv)*s?1 + (z%bdiv)*s?2.
// Epilogue: C = act( (acc*alpha + sbias[z%smod] + bias[m]) [*bn] ) [+resid]
// ---------------------------------------------------------------------------
__global__ void gemm_wmma(const float* __restrict__ A, const float* __restrict__ Bm,
                          float* __restrict__ C, int M, int K,
                          long Asm_, long Ask, long Bsk, long Bsn, long Csm, long Csn,
                          int bdiv, long sA1, long sA2, long sB1, long sB2,
                          long sC1, long sC2,
                          const float* __restrict__ bias, const float* __restrict__ bng,
                          const float* __restrict__ bnb, float bnscale,
                          const float* __restrict__ resid,
                          const float* __restrict__ sbias, int smod,
                          float alpha, int act)
{
  __shared__ __align__(32) _Float16 Al[64 * 32];   // [m][perm(k)]
  __shared__ __align__(32) _Float16 Blp[64 * 32];  // [n][perm(k)]

  int lane = threadIdx.x, w = threadIdx.y;
  int tid  = w * 32 + lane;          // 0..127
  int half = lane >> 4, lr = lane & 15;
  int n0 = blockIdx.x * 64;
  int m0 = blockIdx.y * 64;
  int z = blockIdx.z;
  int bb = z / bdiv, rr = z % bdiv;
  const float* Az = A  + (long)bb * sA1 + (long)rr * sA2;
  const float* Bz = Bm + (long)bb * sB1 + (long)rr * sB2;
  float*       Cz = C  + (long)bb * sC1 + (long)rr * sC2;
  const float* Rz = resid ? (resid + (long)bb * sC1 + (long)rr * sC2) : nullptr;

  int  mw = m0 + w * 16;             // wave-uniform
  bool active = (mw < M);

  v8f acc[4] = {};
  for (int k0 = 0; k0 < K; k0 += 32) {
    // ---- cooperative stage: A 64x32 (K-permuted), B 32x64 (transposed) ----
#pragma unroll
    for (int e = tid; e < 64 * 32; e += 128) {
      int m = e >> 5, k = e & 31;
      int mi = m0 + m; if (mi >= M) mi = M - 1;          // clamp ragged M
      Al[m * 32 + aperm(k)] = (_Float16)Az[(long)mi * Asm_ + (long)(k0 + k) * Ask];
    }
#pragma unroll
    for (int e = tid; e < 32 * 64; e += 128) {
      int k = e >> 6, n = e & 63;                        // coalesced global read
      Blp[n * 32 + aperm(k)] = (_Float16)Bz[(long)(k0 + k) * Bsk + (long)(n0 + n) * Bsn];
    }
    if (k0 + 32 < K) {   // prefetch next K tile (global_prefetch_b8)
      int m = tid >> 5, k = tid & 31;
      int mi = m0 + m; if (mi >= M) mi = M - 1;
      __builtin_prefetch(&Az[(long)mi * Asm_ + (long)(k0 + 32 + k) * Ask], 0, 1);
      __builtin_prefetch(&Bz[(long)(k0 + 32 + (tid & 31)) * Bsk +
                             (long)(n0 + (tid >> 5)) * Bsn], 0, 1);
    }
    __syncthreads();

    if (active) {
      // one contiguous, 32B-aligned fragment per operand
      v16h af = *(const v16h*)&Al[(w * 16 + lr) * 32 + (half << 4)];
#pragma unroll
      for (int t = 0; t < 4; ++t) {
        v16h bf = *(const v16h*)&Blp[(t * 16 + lr) * 32 + (half << 4)];
        acc[t] = __builtin_amdgcn_wmma_f32_16x16x32_f16(false, af, false, bf,
                                                        (short)0, acc[t], false, false);
      }
    }
    __syncthreads();
  }

  if (!active) return;
  float sb = sbias ? sbias[z % smod] : 0.f;
#pragma unroll
  for (int t = 0; t < 4; ++t) {
#pragma unroll
    for (int r = 0; r < 8; ++r) {
      int m = mw + r + (half << 3);
      int n = n0 + t * 16 + lr;
      float x = acc[t][r] * alpha + sb;
      if (bias) x += bias[m];
      if (bng)  x = x * bng[m] * bnscale + bnb[m];
      if (act == 1)      x = fmaxf(x, 0.f);
      else if (act == 2) x = 0.5f * x * (1.f + erff(x * 0.70710678118654752f));
      long ci = (long)m * Csm + (long)n * Csn;
      if (Rz) x += Rz[ci];
      Cz[ci] = x;
    }
  }
}

// ---------------------------------------------------------------------------
// Implicit-GEMM 3x3 conv (pad=1) on 32x32 maps, NCHW; LDS-tiled.
// Block = 128 threads (4 waves). Tile: 64 out-channels x 64 spatial (2 rows).
// Per (tap, K-step): stage W 64x32 (K-permuted) and shifted input patch
// 64x32 (spatial-major), then 16 WMMAs (4 per wave).
// ---------------------------------------------------------------------------
__global__ void conv3x3_wmma(const float* __restrict__ X, const float* __restrict__ Wt,
                             float* __restrict__ Y, int Cin, int Cout, long wBatchStride,
                             const float* __restrict__ bias, const float* __restrict__ bng,
                             const float* __restrict__ bnb, float bnscale, int act)
{
  const int H = 32, Wd = 32, S = 1024;
  __shared__ __align__(32) _Float16 Wl[64 * 32];   // [m][perm(k)]
  __shared__ __align__(32) _Float16 Xl[64 * 32];   // [n][perm(k)]

  int lane = threadIdx.x, w = threadIdx.y;
  int tid  = w * 32 + lane;
  int half = lane >> 4, lr = lane & 15;
  int n0 = blockIdx.x * 64;          // 64 spatial positions = rows ypos, ypos+1
  int m0 = blockIdx.y * 64;          // out-channel base (Cout % 64 == 0)
  int b  = blockIdx.z;
  const float* Xb = X + (long)b * Cin * S;
  const float* Wb = Wt + (long)b * wBatchStride;
  float*       Yb = Y + (long)b * Cout * S;
  int ypos = n0 >> 5;

  v8f acc[4] = {};
  for (int ky = 0; ky < 3; ++ky) {
    for (int kx = 0; kx < 3; ++kx) {
      int tap = ky * 3 + kx;
      for (int c0 = 0; c0 < Cin; c0 += 32) {
        // ---- stage weights 64x32 (K-permuted) ----
#pragma unroll
        for (int e = tid; e < 64 * 32; e += 128) {
          int m = e >> 5, k = e & 31;
          Wl[m * 32 + aperm(k)] =
              (_Float16)Wb[(long)(m0 + m) * Cin * 9 + (long)(c0 + k) * 9 + tap];
        }
        // ---- stage shifted input patch (zero-padded), spatial-major ----
#pragma unroll
        for (int e = tid; e < 32 * 64; e += 128) {
          int k = e >> 6, n = e & 63;                    // coalesced along x
          int yy = ypos + (n >> 5) + ky - 1;
          int xx = (n & 31) + kx - 1;
          bool in = (yy >= 0) && (yy < H) && (xx >= 0) && (xx < Wd);
          Xl[n * 32 + aperm(k)] = in ? (_Float16)Xb[(long)(c0 + k) * S + (yy << 5) + xx]
                              : (_Float16)0.f;
        }
        __syncthreads();

        v16h af = *(const v16h*)&Wl[(w * 16 + lr) * 32 + (half << 4)];
#pragma unroll
        for (int t = 0; t < 4; ++t) {
          v16h bf = *(const v16h*)&Xl[(t * 16 + lr) * 32 + (half << 4)];
          acc[t] = __builtin_amdgcn_wmma_f32_16x16x32_f16(false, af, false, bf,
                                                          (short)0, acc[t], false, false);
        }
        __syncthreads();
      }
    }
  }

#pragma unroll
  for (int t = 0; t < 4; ++t) {
#pragma unroll
    for (int r = 0; r < 8; ++r) {
      int m = m0 + w * 16 + r + (half << 3);
      int n = n0 + t * 16 + lr;
      float x = acc[t][r];
      if (bias) x += bias[m];
      if (bng)  x = x * bng[m] * bnscale + bnb[m];
      if (act == 1) x = fmaxf(x, 0.f);
      Yb[(long)m * S + n] = x;
    }
  }
}

// ------------------------- auxiliary (VALU) kernels -------------------------

__global__ void gate_kernel(const float* __restrict__ zp, const float* __restrict__ za,
                            const float* __restrict__ aw, const float* __restrict__ ab,
                            float* __restrict__ w0, int total)
{
  int i = blockIdx.x * blockDim.x + threadIdx.x;
  if (i >= total) return;
  int b = i >> 10, s = i & 1023;
  const float* zpb = zp + (long)b * 256 * 1024;
  const float* zab = za + (long)b * 256 * 1024;
  float l0 = ab[0], l1 = ab[1];
  for (int c = 0; c < 256; ++c) {
    float vp = zpb[(long)c * 1024 + s];
    float va = zab[(long)c * 1024 + s];
    l0 += aw[c] * vp + aw[256 + c] * va;
    l1 += aw[512 + c] * vp + aw[768 + c] * va;
  }
  float m = fmaxf(l0, l1);
  float e0 = expf(l0 - m), e1 = expf(l1 - m);
  w0[i] = e0 / (e0 + e1);
}

__global__ void fuse_kernel(const float* __restrict__ zp, const float* __restrict__ za,
                            const float* __restrict__ w0, float* __restrict__ out, long total)
{
  long i = (long)blockIdx.x * blockDim.x + threadIdx.x;
  if (i >= total) return;
  long b = i >> 18;
  int  s = (int)(i & 1023);
  float w = w0[(b << 10) | s];
  out[i] = w * zp[i] + (1.f - w) * za[i];
}

__global__ void gap_kernel(const float* __restrict__ X, float* __restrict__ out, int total)
{
  int i = blockIdx.x * blockDim.x + threadIdx.x;
  if (i >= total) return;
  const float* p = X + (long)i * 1024;
  float s = 0.f;
  for (int j = 0; j < 1024; ++j) s += p[j];
  out[i] = s * (1.f / 1024.f);
}

__global__ void fc_relu_kernel(const float* __restrict__ in, const float* __restrict__ w,
                               const float* __restrict__ bias, float* __restrict__ out,
                               int In, int Out, int total)
{
  int i = blockIdx.x * blockDim.x + threadIdx.x;
  if (i >= total) return;
  int b = i / Out, o = i % Out;
  float s = bias[o];
  for (int k = 0; k < In; ++k) s += in[(long)b * In + k] * w[(long)o * In + k];
  out[i] = fmaxf(s, 0.f);
}

__global__ void mix_kernel(const float* __restrict__ h, const float* __restrict__ w,
                           const float* __restrict__ bias, float* __restrict__ mix)
{
  int b = threadIdx.x;
  if (b >= 8) return;
  float l[4];
  float m = -3.4e38f;
  for (int n = 0; n < 4; ++n) {
    float s = bias[n];
    for (int k = 0; k < 64; ++k) s += w[n * 64 + k] * h[b * 64 + k];
    l[n] = s; m = fmaxf(m, s);
  }
  float t = 0.f;
  for (int n = 0; n < 4; ++n) { l[n] = expf(l[n] - m); t += l[n]; }
  float inv = 1.f / t;
  for (int n = 0; n < 4; ++n) mix[b * 4 + n] = l[n] * inv;
}

__global__ void mix_bank_kernel(const float* __restrict__ mix, const float* __restrict__ bank,
                                float* __restrict__ dk, long perB, long total)
{
  for (long i = (long)blockIdx.x * blockDim.x + threadIdx.x; i < total;
       i += (long)gridDim.x * blockDim.x) {
    long b = i / perB, e = i % perB;
    float s = 0.f;
#pragma unroll
    for (int n = 0; n < 4; ++n) s += mix[b * 4 + n] * bank[(long)n * perB + e];
    dk[i] = s;
  }
}

__global__ void gn_kernel(const float* __restrict__ X, const float* __restrict__ g,
                          const float* __restrict__ bta, float* __restrict__ Y)
{
  __shared__ float s1[256], s2[256];
  int b = blockIdx.x >> 5, grp = blockIdx.x & 31;
  long base = ((long)b * 256 + grp * 8) * 1024;
  const float* Xb = X + base;
  float* Yb = Y + base;
  int tid = threadIdx.x;
  float a = 0.f, q = 0.f;
  for (int i = tid; i < 8192; i += 256) { float v = Xb[i]; a += v; q += v * v; }
  s1[tid] = a; s2[tid] = q; __syncthreads();
  for (int st = 128; st > 0; st >>= 1) {
    if (tid < st) { s1[tid] += s1[tid + st]; s2[tid] += s2[tid + st]; }
    __syncthreads();
  }
  float mean = s1[0] * (1.f / 8192.f);
  float var  = s2[0] * (1.f / 8192.f) - mean * mean;
  float inv = rsqrtf(var + 1e-5f);
  for (int i = tid; i < 8192; i += 256) {
    int c = grp * 8 + (i >> 10);
    Yb[i] = (Xb[i] - mean) * inv * g[c] + bta[c];
  }
}

__global__ void softmax_kernel(float* __restrict__ X, int L)
{
  __shared__ float sm[256];
  float* row = X + (long)blockIdx.x * L;
  int tid = threadIdx.x;
  float m = -3.4e38f;
  for (int i = tid; i < L; i += 256) m = fmaxf(m, row[i]);
  sm[tid] = m; __syncthreads();
  for (int st = 128; st > 0; st >>= 1) {
    if (tid < st) sm[tid] = fmaxf(sm[tid], sm[tid + st]);
    __syncthreads();
  }
  float M = sm[0]; __syncthreads();
  float s = 0.f;
  for (int i = tid; i < L; i += 256) { float e = expf(row[i] - M); row[i] = e; s += e; }
  sm[tid] = s; __syncthreads();
  for (int st = 128; st > 0; st >>= 1) {
    if (tid < st) sm[tid] += sm[tid + st];
    __syncthreads();
  }
  float inv = 1.f / sm[0];
  for (int i = tid; i < L; i += 256) row[i] *= inv;
}

__global__ void final1x1_kernel(const float* __restrict__ X, const float* __restrict__ w,
                                const float* __restrict__ b2, float* __restrict__ out, int total)
{
  int i = blockIdx.x * blockDim.x + threadIdx.x;
  if (i >= total) return;
  int b = i >> 10, s = i & 1023;
  const float* Xb = X + (long)b * 128 * 1024;
  float acc = b2[0];
  for (int c = 0; c < 128; ++c) acc += w[c] * Xb[(long)c * 1024 + s];
  out[i] = acc;
}

// ------------------------------ host driver --------------------------------

static inline void GEMM(hipStream_t st, const float* A, const float* B, float* C,
                        int M, int N, int K,
                        long Asm_, long Ask, long Bsk, long Bsn, long Csm, long Csn,
                        int Z, long sA1, long sB1, long sC1,
                        const float* bias, const float* bng, const float* bnb, float bnscale,
                        const float* resid, const float* sbias, int smod,
                        float alpha, int act)
{
  dim3 blk(32, 4), grd(N / 64, (M + 63) / 64, Z);
  gemm_wmma<<<grd, blk, 0, st>>>(A, B, C, M, K, Asm_, Ask, Bsk, Bsn, Csm, Csn,
                                 /*bdiv=*/1, sA1, 0, sB1, 0, sC1, 0,
                                 bias, bng, bnb, bnscale, resid, sbias, smod, alpha, act);
}

extern "C" void kernel_launch(void* const* d_in, const int* in_sizes, int n_in,
                              void* d_out, int out_size, void* d_ws, size_t ws_size,
                              hipStream_t stream)
{
  (void)in_sizes; (void)n_in; (void)out_size; (void)ws_size;
  auto F = [&](int i) { return (const float*)d_in[i]; };

  const int  B = 8, C = 256, S = 1024, NH = 8;
  const long MAP = (long)B * C * S;
  const float bnc = 1.0f / sqrtf(1.0f + 1e-5f);   // eval-BN scale factor

  float* ws = (float*)d_ws;
  long off = 0;
  auto alloc = [&](long n) { float* p = ws + off; off += n; return p; };
  float* X0   = alloc(MAP);
  float* X1   = alloc(MAP);
  float* X2   = alloc(MAP);
  float* GB   = alloc(MAP);
  float* QKV  = alloc((long)B * 3 * C * S);
  float* ATT  = alloc((long)NH * S * S);
  float* OB   = alloc(MAP);
  float* HB   = alloc((long)B * 1024 * S);
  float* DK   = alloc((long)B * C * C * 9);
  float* W0B  = alloc((long)B * S);
  float* GAPB = alloc((long)B * C);
  float* H1B  = alloc((long)B * 64);
  float* MIXB = alloc((long)B * 4);
  float* H128 = alloc((long)B * 128 * S);

  // ---------------- feature fusion ----------------
  GEMM(stream, F(2), F(0), X0, 256, 1024, 256, 256, 1, (long)S, 1, (long)S, 1,
       B, 0, (long)C * S, (long)C * S, F(3), F(4), F(5), bnc,
       nullptr, nullptr, 1, 1.f, 1);
  GEMM(stream, F(6), F(1), X1, 256, 1024, 256, 256, 1, (long)S, 1, (long)S, 1,
       B, 0, (long)C * S, (long)C * S, F(7), F(8), F(9), bnc,
       nullptr, nullptr, 1, 1.f, 1);
  gate_kernel<<<(B * S + 255) / 256, 256, 0, stream>>>(X0, X1, F(10), F(11), W0B, B * S);
  fuse_kernel<<<(int)((MAP + 255) / 256), 256, 0, stream>>>(X0, X1, W0B, GB, MAP);
  {
    dim3 g(16, 4, 8), blk(32, 4);
    conv3x3_wmma<<<g, blk, 0, stream>>>(GB, F(12), X2, 256, 256, 0,
                                        F(13), F(14), F(15), bnc, 1);
  }
  float* cur = X2; float* alt = X0;

  // ---------------- dynamic convs x3 ----------------
  for (int d = 0; d < 3; ++d) {
    int p = 16 + 8 * d;
    long perB = (long)C * C * 9;
    gap_kernel<<<(B * C + 255) / 256, 256, 0, stream>>>(cur, GAPB, B * C);
    fc_relu_kernel<<<(B * 64 + 255) / 256, 256, 0, stream>>>(GAPB, F(p), F(p + 1), H1B,
                                                             256, 64, B * 64);
    mix_kernel<<<1, 32, 0, stream>>>(H1B, F(p + 2), F(p + 3), MIXB);
    mix_bank_kernel<<<4096, 256, 0, stream>>>(MIXB, F(p + 4), DK, perB, (long)B * perB);
    dim3 g(16, 4, 8), blk(32, 4);
    conv3x3_wmma<<<g, blk, 0, stream>>>(cur, DK, alt, 256, 256, perB,
                                        F(p + 5), F(p + 6), F(p + 7), bnc, 1);
    float* t = cur; cur = alt; alt = t;
  }

  // ---------------- transformer blocks x4 ----------------
  for (int t4 = 0; t4 < 4; ++t4) {
    int p = 40 + 12 * t4;
    gn_kernel<<<256, 256, 0, stream>>>(cur, F(p), F(p + 1), GB);
    GEMM(stream, F(p + 4), GB, QKV, 768, 1024, 256, 256, 1, (long)S, 1, (long)S, 1,
         B, 0, (long)C * S, (long)3 * C * S, nullptr, nullptr, nullptr, 1.f,
         nullptr, nullptr, 1, 1.f, 0);
    for (int b = 0; b < B; ++b) {
      const float* qb = QKV + (long)b * 3 * C * S;
      // scores: Q^T K * 1/sqrt(hd) + pbias[h]   (z = head)
      GEMM(stream, qb, qb + (long)256 * S, ATT, 1024, 1024, 32,
           1, (long)S, (long)S, 1, (long)S, 1,
           NH, (long)32 * S, (long)32 * S, (long)S * S,
           nullptr, nullptr, nullptr, 1.f, nullptr, F(p + 7), NH,
           0.17677669529663689f, 0);
      softmax_kernel<<<NH * S, 256, 0, stream>>>(ATT, S);
      // out[h*32+d, s] = sum_t V[d,t] * attn[s,t]
      GEMM(stream, qb + (long)512 * S, ATT, OB + (long)b * C * S, 32, 1024, 1024,
           (long)S, 1, 1, (long)S, (long)S, 1,
           NH, (long)32 * S, (long)S * S, (long)32 * S,
           nullptr, nullptr, nullptr, 1.f, nullptr, nullptr, 1, 1.f, 0);
    }
    // proj + residual (in place into cur)
    GEMM(stream, F(p + 5), OB, cur, 256, 1024, 256, 256, 1, (long)S, 1, (long)S, 1,
         B, 0, (long)C * S, (long)C * S, F(p + 6), nullptr, nullptr, 1.f,
         cur, nullptr, 1, 1.f, 0);
    // MLP
    gn_kernel<<<256, 256, 0, stream>>>(cur, F(p + 2), F(p + 3), GB);
    GEMM(stream, F(p + 8), GB, HB, 1024, 1024, 256, 256, 1, (long)S, 1, (long)S, 1,
         B, 0, (long)C * S, (long)1024 * S, F(p + 9), nullptr, nullptr, 1.f,
         nullptr, nullptr, 1, 1.f, 2);
    GEMM(stream, F(p + 10), HB, cur, 256, 1024, 1024, 1024, 1, (long)S, 1, (long)S, 1,
         B, 0, (long)1024 * S, (long)C * S, F(p + 11), nullptr, nullptr, 1.f,
         cur, nullptr, 1, 1.f, 0);
  }

  // ---------------- refine convs x3 ----------------
  for (int r = 0; r < 3; ++r) {
    int p = 88 + 4 * r;
    dim3 g(16, 4, 8), blk(32, 4);
    conv3x3_wmma<<<g, blk, 0, stream>>>(cur, F(p), alt, 256, 256, 0,
                                        F(p + 1), F(p + 2), F(p + 3), bnc, 1);
    float* t = cur; cur = alt; alt = t;
  }

  // ---------------- final head ----------------
  {
    dim3 g(16, 2, 8), blk(32, 4);
    conv3x3_wmma<<<g, blk, 0, stream>>>(cur, F(100), H128, 256, 128, 0,
                                        F(101), F(102), F(103), bnc, 1);
  }
  final1x1_kernel<<<(B * S + 255) / 256, 256, 0, stream>>>(H128, F(104), F(105),
                                                           (float*)d_out, B * S);
}
